// LRSelfAttentionHead_17463337026244
// MI455X (gfx1250) — matrix operand order repaired and checked by reading, not compiled, not run on hardware
//
#include <hip/hip_runtime.h>

typedef __attribute__((ext_vector_type(16))) _Float16 v16h;
typedef __attribute__((ext_vector_type(8)))  _Float16 v8h;
typedef __attribute__((ext_vector_type(8)))  float    v8f;
typedef __attribute__((ext_vector_type(4)))  float    v4f;
typedef __attribute__((ext_vector_type(4)))  unsigned int v4u;
typedef __attribute__((ext_vector_type(4)))  int      v4i;
typedef __attribute__((ext_vector_type(8)))  int      v8i;

#ifndef __has_builtin
#define __has_builtin(x) 0
#endif
#if __has_builtin(__builtin_amdgcn_tensor_load_to_lds) && __has_builtin(__builtin_amdgcn_s_wait_tensorcnt)
#define USE_TDM 1
#else
#define USE_TDM 0
#endif

#define DIM   1024
#define RANK  32
#define BATCH 4
#define SEQ   4096
#define ROWS  (BATCH*SEQ)   // 16384 flat (b,t) rows

#define UT    32            // u-tile == K of one f16 WMMA
#define BST   40            // f16 row stride of TDM-staged B tiles (32 + 4-dword pad)
#define PLD   40            // padded LDS stride for p tile
#define CLD   1032          // padded LDS stride for ctx tile (1024+8)
#define BUFE  (DIM*BST)     // f16 elements per staging buffer (81920 B)

union AFrag { v16h v; v8h h[2]; };

__device__ __forceinline__ v8f wmma_f16(v16h a, v16h b, v8f c) {
  return __builtin_amdgcn_wmma_f32_16x16x32_f16(false, a, false, b, (short)0, c, false, false);
}

// A fragment (16x32 f16): element (m,k) = base[(row0+m)*ld + k0 + k]
__device__ __forceinline__ v16h load_a_f16(const _Float16* base, int ld, int row0, int k0, int lane) {
  int m = lane & 15, hi = lane >> 4;
  const _Float16* p = base + (size_t)(row0 + m) * ld + k0;
  AFrag f;
  f.h[0] = *(const v8h*)(p + 8*hi);
  f.h[1] = *(const v8h*)(p + 16 + 8*hi);
  return f.v;
}

// B fragment (32x16 f16): element (k,n) = baseT[(n0+n)*ld + k0 + k]
__device__ __forceinline__ v16h load_b_f16(const _Float16* baseT, int ld, int n0, int k0, int lane) {
  int n = lane & 15, hi = lane >> 4;
  const _Float16* p = baseT + (size_t)(n0 + n) * ld + k0 + 8*hi;
  AFrag f;
  f.h[0] = *(const v8h*)(p);
  f.h[1] = *(const v8h*)(p + 16);
  return f.v;
}

#if USE_TDM
// TDM: DMA a 2D tile (tile_d0 contiguous elems x tile_d1 lines, line stride stride0
// elements, 2-byte elements) from global to LDS at byte offset lds_off, inserting a
// 4-dword pad after each 16-dword (one 32-elem row) burst -> LDS row stride = BST f16.
__device__ __forceinline__ void tdm_load_2d(unsigned lds_off, const void* gptr,
                                            unsigned tensor_d0, unsigned tensor_d1,
                                            unsigned long long stride0,
                                            unsigned tile_d0, unsigned tile_d1) {
  unsigned long long ga = (unsigned long long)(__UINTPTR_TYPE__)gptr;
  v4u g0;
  g0[0] = 1u;                                            // count=1, no gather
  g0[1] = lds_off;                                       // lds_addr (bytes)
  g0[2] = (unsigned)(ga & 0xffffffffu);                  // global_addr[31:0]
  g0[3] = (unsigned)((ga >> 32) & 0x1ffffffu) | (2u << 30);  // addr[56:32] | type=2
  v8i g1;
  g1[0] = (int)((1u << 16) | (1u << 20) | (3u << 22) | (3u << 25)); // data_size=2B, pad: 16dw interval, 4dw amount
  g1[1] = (int)((tensor_d0 & 0xffffu) << 16);            // tensor_dim0[15:0] (barrier addr 0)
  g1[2] = (int)((tensor_d0 >> 16) | ((tensor_d1 & 0xffffu) << 16));
  g1[3] = (int)((tensor_d1 >> 16) | (tile_d0 << 16));    // tensor_dim1[31:16] | tile_dim0
  g1[4] = (int)(tile_d1 & 0xffffu);                      // tile_dim1 (tile_dim2=0)
  g1[5] = (int)(stride0 & 0xffffffffu);                  // tensor_dim0_stride[31:0]
  g1[6] = (int)((stride0 >> 32) & 0xffffu);              // stride[47:32] (dim1_stride=0)
  g1[7] = 0;
  v4i z = {0, 0, 0, 0};
#if __clang_major__ >= 23
  v8i z8 = {0, 0, 0, 0, 0, 0, 0, 0};
  __builtin_amdgcn_tensor_load_to_lds(g0, g1, z, z, z8, 0);
#else
  __builtin_amdgcn_tensor_load_to_lds(g0, g1, z, z, 0);
#endif
}
#endif

// ---------- prep: tiled transpose + f32->f16 convert: dst[c][r] = src[r][c] ----------
__global__ void k_transpose_f16(const float* __restrict__ src, _Float16* __restrict__ dst,
                                int R, int C) {
  __shared__ _Float16 tile[32][33];
  int rt = blockIdx.x % (R >> 5);
  int ct = blockIdx.x / (R >> 5);
  int r0 = rt * 32, c0 = ct * 32;
  int j = threadIdx.x & 31, i0 = threadIdx.x >> 5;
#pragma unroll
  for (int i = i0; i < 32; i += 8)
    tile[i][j] = (_Float16)src[(size_t)(r0 + i) * C + c0 + j];
  __syncthreads();
#pragma unroll
  for (int i = i0; i < 32; i += 8)
    dst[(size_t)(c0 + i) * R + r0 + j] = tile[j][i];
}

__global__ void k_qk_cvt(const float* __restrict__ Q, const float* __restrict__ K,
                         _Float16* __restrict__ Qh, _Float16* __restrict__ Kh) {
  int i = blockIdx.x * 256 + threadIdx.x;   // < RANK*DIM
  Qh[i] = (_Float16)Q[i];
  Kh[i] = (_Float16)K[i];
}

// ---------- projections: xq = x.Q^T * (1/sqrt(d)), xk = x.K^T ----------
__global__ void k_project(const float* __restrict__ x,
                          const _Float16* __restrict__ Qh, const _Float16* __restrict__ Kh,
                          _Float16* __restrict__ xqh, _Float16* __restrict__ xkh) {
  int wave = (blockIdx.x * blockDim.x + threadIdx.x) >> 5;
  int lane = threadIdx.x & 31;
  int t0 = wave * 16;
  int m = lane & 15, hi = lane >> 4;
  v8f aq0 = {}, aq1 = {}, ak0 = {}, ak1 = {};
  for (int d0 = 0; d0 < DIM; d0 += 32) {
    const float* px = x + (size_t)(t0 + m) * DIM + d0;
    v4f f0 = *(const v4f*)(px + 8*hi);
    v4f f1 = *(const v4f*)(px + 8*hi + 4);
    v4f f2 = *(const v4f*)(px + 16 + 8*hi);
    v4f f3 = *(const v4f*)(px + 16 + 8*hi + 4);
    AFrag a;
#pragma unroll
    for (int e = 0; e < 4; e++) {
      a.h[0][e]   = (_Float16)f0[e];  a.h[0][4+e] = (_Float16)f1[e];
      a.h[1][e]   = (_Float16)f2[e];  a.h[1][4+e] = (_Float16)f3[e];
    }
    v16h bq0 = load_b_f16(Qh, DIM, 0,  d0, lane);
    v16h bq1 = load_b_f16(Qh, DIM, 16, d0, lane);
    v16h bk0 = load_b_f16(Kh, DIM, 0,  d0, lane);
    v16h bk1 = load_b_f16(Kh, DIM, 16, d0, lane);
    aq0 = wmma_f16(a.v, bq0, aq0);
    aq1 = wmma_f16(a.v, bq1, aq1);
    ak0 = wmma_f16(a.v, bk0, ak0);
    ak1 = wmma_f16(a.v, bk1, ak1);
  }
  const float scale = 0.03125f;  // 1/sqrt(1024)
#pragma unroll
  for (int r = 0; r < 8; r++) {
    int row = t0 + r + 8*hi;
    xqh[(size_t)row * RANK + m]      = (_Float16)(aq0[r] * scale);
    xqh[(size_t)row * RANK + 16 + m] = (_Float16)(aq1[r] * scale);
    xkh[(size_t)row * RANK + m]      = (_Float16)ak0[r];
    xkh[(size_t)row * RANK + 16 + m] = (_Float16)ak1[r];
  }
}

// ---------- pass 1: per-row softmax stats (max, 1/sumexp) ----------
__global__ void k_stats(const _Float16* __restrict__ xqh, const _Float16* __restrict__ xkh,
                        float* __restrict__ rowmax, float* __restrict__ rsinv) {
  __shared__ float mW[8][16], sW[8][16];
  int w = threadIdx.x >> 5, lane = threadIdx.x & 31;
  int t0 = blockIdx.x * 16;
  int b  = t0 / SEQ;
  int hi = lane >> 4;
  v16h a = load_a_f16(xqh, RANK, t0, 0, lane);
  float mr[8], sr[8];
#pragma unroll
  for (int r = 0; r < 8; r++) { mr[r] = -1e30f; sr[r] = 0.f; }
  v8f zc = {};
  for (int u0 = w * 16; u0 < SEQ; u0 += 128) {
    v16h bm = load_b_f16(xkh, RANK, b * SEQ + u0, 0, lane);
    v8f d = wmma_f16(a, bm, zc);
#pragma unroll
    for (int r = 0; r < 8; r++) {
      float v  = d[r];
      float nm = fmaxf(mr[r], v);
      sr[r] = sr[r] * __expf(mr[r] - nm) + __expf(v - nm);
      mr[r] = nm;
    }
  }
#pragma unroll
  for (int off = 8; off >= 1; off >>= 1) {
#pragma unroll
    for (int r = 0; r < 8; r++) {
      float mo = __shfl_xor(mr[r], off, 16);
      float so = __shfl_xor(sr[r], off, 16);
      float nm = fmaxf(mr[r], mo);
      sr[r] = sr[r] * __expf(mr[r] - nm) + so * __expf(mo - nm);
      mr[r] = nm;
    }
  }
  if ((lane & 15) == 0) {
#pragma unroll
    for (int r = 0; r < 8; r++) { mW[w][r + 8*hi] = mr[r]; sW[w][r + 8*hi] = sr[r]; }
  }
  __syncthreads();
  if (threadIdx.x < 16) {
    int row = threadIdx.x;
    float M = -1e30f, S = 0.f;
#pragma unroll
    for (int wv = 0; wv < 8; wv++) {
      float mo = mW[wv][row], so = sW[wv][row];
      float nm = fmaxf(M, mo);
      S = S * __expf(M - nm) + so * __expf(mo - nm);
      M = nm;
    }
    rowmax[t0 + row] = M;
    rsinv[t0 + row]  = 1.0f / S;
  }
}

// ---------- pass 2: ctx = softmax(scores).x, fused out = ctx.VO ----------
__global__ void __launch_bounds__(256, 1)
k_ctxout(const _Float16* __restrict__ xqh, const _Float16* __restrict__ xkh,
         const _Float16* __restrict__ xhT, const _Float16* __restrict__ VOT,
         const float* __restrict__ rowmax, const float* __restrict__ rsinv,
         float* __restrict__ out) {
  extern __shared__ __align__(16) unsigned char smem[];
  _Float16* bbuf = (_Float16*)smem;                                  // 2 x [DIM][BST] staged B tiles
  _Float16* pbuf = (_Float16*)(smem + 2 * BUFE * 2);                 // [32][PLD]
  _Float16* ctx  = (_Float16*)(smem + 2 * BUFE * 2 + 32 * PLD * 2);  // [32][CLD]
  float*    mrow = (float*)(smem + 2 * BUFE * 2 + 32 * PLD * 2 + 32 * CLD * 2);
  float*    rinv = mrow + 32;

  int w = threadIdx.x >> 5, lane = threadIdx.x & 31;
  int t0 = blockIdx.x * 32;
  int b  = t0 / SEQ;
  int m  = lane & 15, hi = lane >> 4;
  int rb = w & 1;                 // 16-row block
  int colbase = (w >> 1) * 256;   // 256 d-columns per wave
  int ub = w >> 1;                // score u-sub-block (valid for w<4)

  if (threadIdx.x < 32) {
    mrow[threadIdx.x] = rowmax[t0 + threadIdx.x];
    rinv[threadIdx.x] = rsinv[t0 + threadIdx.x];
  }

  v16h aq = load_a_f16(xqh, RANK, t0 + rb * 16, 0, lane);  // K=32 == RANK
  v8f zc = {};
  v8f acc[16];
#pragma unroll
  for (int i = 0; i < 16; i++) acc[i] = zc;

  const int NU = SEQ / UT;
#if USE_TDM
  if (w == 0) {
    tdm_load_2d(0, xhT + (size_t)b * SEQ, ROWS, DIM, (unsigned long long)ROWS, UT, DIM);
    __builtin_amdgcn_s_wait_tensorcnt((short)0);
  }
#endif
  __syncthreads();

  for (int it = 0; it < NU; ++it) {
    int u0 = it * UT;
    _Float16* buf = bbuf + (it & 1) * BUFE;
#if USE_TDM
    if (w == 0 && it + 1 < NU)   // DMA next tile while computing this one
      tdm_load_2d((unsigned)(((it + 1) & 1) * BUFE * 2),
                  xhT + (size_t)b * SEQ + (u0 + UT), ROWS, DIM,
                  (unsigned long long)ROWS, UT, DIM);
#endif
    if (w < 4) {  // scores for one 16x16 block -> probabilities -> LDS
      v16h bk = load_b_f16(xkh, RANK, b * SEQ + u0 + ub * 16, 0, lane);
      v8f sc = wmma_f16(aq, bk, zc);
#pragma unroll
      for (int r = 0; r < 8; r++) {
        int lrow = rb * 16 + r + 8 * hi;
        float p = __expf(sc[r] - mrow[lrow]) * rinv[lrow];
        pbuf[lrow * PLD + ub * 16 + m] = (_Float16)p;
      }
    }
    __syncthreads();
    // ctx += p @ x
    v16h ap = load_a_f16(pbuf, PLD, rb * 16, 0, lane);
#pragma unroll
    for (int ct = 0; ct < 16; ct++) {
#if USE_TDM
      v16h bx = load_b_f16(buf, BST, colbase + ct * 16, 0, lane);        // from LDS
#else
      v16h bx = load_b_f16(xhT, ROWS, colbase + ct * 16, b * SEQ + u0, lane);
#endif
      acc[ct] = wmma_f16(ap, bx, acc[ct]);
    }
#if USE_TDM
    if (w == 0 && it + 1 < NU) __builtin_amdgcn_s_wait_tensorcnt((short)0);
#endif
    __syncthreads();
  }

  // stage ctx (f16) in LDS for the fused output projection
#pragma unroll
  for (int ct = 0; ct < 16; ct++)
#pragma unroll
    for (int r = 0; r < 8; r++)
      ctx[(rb * 16 + r + 8 * hi) * CLD + colbase + ct * 16 + m] = (_Float16)acc[ct][r];
#pragma unroll
  for (int i = 0; i < 16; i++) acc[i] = zc;

#if USE_TDM
  if (w == 0) {
    tdm_load_2d(0, VOT, DIM, DIM, (unsigned long long)DIM, UT, DIM);
    __builtin_amdgcn_s_wait_tensorcnt((short)0);
  }
#endif
  __syncthreads();

  // out = ctx @ VO
  int ebase = colbase;
  for (int kk = 0; kk < DIM / 32; ++kk) {
    _Float16* buf = bbuf + (kk & 1) * BUFE;
#if USE_TDM
    if (w == 0 && kk + 1 < DIM / 32)
      tdm_load_2d((unsigned)(((kk + 1) & 1) * BUFE * 2),
                  VOT + (kk + 1) * 32, DIM, DIM, (unsigned long long)DIM, UT, DIM);
#endif
    v16h ac = load_a_f16(ctx, CLD, rb * 16, kk * 32, lane);
#pragma unroll
    for (int ct = 0; ct < 16; ct++) {
#if USE_TDM
      v16h bv = load_b_f16(buf, BST, ebase + ct * 16, 0, lane);          // from LDS
#else
      v16h bv = load_b_f16(VOT, DIM, ebase + ct * 16, kk * 32, lane);
#endif
      acc[ct] = wmma_f16(ac, bv, acc[ct]);
    }
#if USE_TDM
    if (w == 0 && kk + 1 < DIM / 32) __builtin_amdgcn_s_wait_tensorcnt((short)0);
#endif
    __syncthreads();
  }
#pragma unroll
  for (int ct = 0; ct < 16; ct++)
#pragma unroll
    for (int r = 0; r < 8; r++)
      out[(size_t)(t0 + rb * 16 + r + 8 * hi) * DIM + ebase + ct * 16 + m] = acc[ct][r];
}

extern "C" void kernel_launch(void* const* d_in, const int* in_sizes, int n_in,
                              void* d_out, int out_size, void* d_ws, size_t ws_size,
                              hipStream_t stream) {
  (void)in_sizes; (void)n_in; (void)out_size; (void)ws_size;
  const float* x  = (const float*)d_in[0];
  const float* Q  = (const float*)d_in[1];
  const float* K  = (const float*)d_in[2];
  const float* VO = (const float*)d_in[3];
  float* out = (float*)d_out;

  char* ws = (char*)d_ws;
  _Float16* xhT = (_Float16*)ws; ws += (size_t)ROWS * DIM * 2;   // 32 MB  x^T f16
  _Float16* VOT = (_Float16*)ws; ws += (size_t)DIM * DIM * 2;    // 2 MB   VO^T f16
  _Float16* Qh  = (_Float16*)ws; ws += (size_t)RANK * DIM * 2;
  _Float16* Kh  = (_Float16*)ws; ws += (size_t)RANK * DIM * 2;
  _Float16* xqh = (_Float16*)ws; ws += (size_t)ROWS * RANK * 2;  // 1 MB
  _Float16* xkh = (_Float16*)ws; ws += (size_t)ROWS * RANK * 2;  // 1 MB
  float* rowmax = (float*)ws;    ws += (size_t)ROWS * 4;
  float* rsinv  = (float*)ws;    ws += (size_t)ROWS * 4;

  k_transpose_f16<<<(ROWS / 32) * (DIM / 32), 256, 0, stream>>>(x, xhT, ROWS, DIM);
  k_transpose_f16<<<(DIM / 32) * (DIM / 32), 256, 0, stream>>>(VO, VOT, DIM, DIM);
  k_qk_cvt<<<(RANK * DIM) / 256, 256, 0, stream>>>(Q, K, Qh, Kh);
  k_project<<<(ROWS / 16) / 8, 256, 0, stream>>>(x, Qh, Kh, xqh, xkh);
  k_stats<<<ROWS / 16, 256, 0, stream>>>(xqh, xkh, rowmax, rsinv);
  size_t smem = (size_t)2 * BUFE * 2 + (size_t)32 * PLD * 2 + (size_t)32 * CLD * 2 + 64 * 4;
  k_ctxout<<<ROWS / 32, 256, smem, stream>>>(xqh, xkh, xhT, VOT, rowmax, rsinv, out);
}
